// LFM3BDecoderLayer_63531156242853
// MI455X (gfx1250) — matrix operand order repaired and checked by reading, not compiled, not run on hardware
//
#include <hip/hip_runtime.h>

// ---------------- problem constants (match reference) ----------------
constexpr int kH   = 32;     // query heads
constexpr int kKVH = 8;      // kv heads
constexpr int kD   = 64;     // head dim
constexpr int kHID = 2048;   // hidden
constexpr int kFF  = 8192;   // ffn
constexpr int kB   = 2;
constexpr int kS   = 2048;
constexpr int kM   = kB * kS;          // 4096 tokens
constexpr float kEPS = 1e-6f;

// ---------------- types ----------------
typedef __bf16 bf16;
typedef bf16  v16bf __attribute__((ext_vector_type(16)));
typedef float v8f   __attribute__((ext_vector_type(8)));

__device__ inline v8f wmma_bf16(v16bf a, v16bf b, v8f c) {
  // D = A(16x32 bf16) * B(32x16 bf16) + C(16x16 f32)
  return __builtin_amdgcn_wmma_f32_16x16x32_bf16(
      /*neg_a=*/false, a, /*neg_b=*/false, b,
      /*c_mod=*/(short)0, c, /*reuse_a=*/false, /*reuse_b=*/false);
}

// A-matrix fragment (16-bit A 16x32, ISA 7.12.2):
// lane m = lane&15, half = lane>>4.
// VGPR0..3 : K = half*8 + 2v, 2v+1       (covers K 0..15)
// VGPR4..7 : K = 16 + half*8 + 2v, 2v+1  (covers K 16..31)
__device__ inline v16bf load_a_frag(const bf16* base, int stride, int lane, int kb) {
  const int m = lane & 15, half = lane >> 4;
  const bf16* p = base + m * stride;
  v16bf a;
#pragma unroll
  for (int v = 0; v < 4; ++v) {
    int k0 = kb + half * 8 + 2 * v;
    a[2 * v]     = p[k0];
    a[2 * v + 1] = p[k0 + 1];
    int k1 = kb + 16 + half * 8 + 2 * v;
    a[8 + 2 * v]     = p[k1];
    a[8 + 2 * v + 1] = p[k1 + 1];
  }
  return a;
}

// B-matrix fragment (16-bit B 32x16), reading from B^T stored [N][K]:
// lane n = lane&15, half = lane>>4; VGPR v<4: K = half*8 + 2v, 2v+1; v>=4: K = 16 + half*8 + 2(v-4), +1.
__device__ inline v16bf load_b_frag(const bf16* baseT, int stride, int lane, int kb) {
  const int n = lane & 15, half = lane >> 4;
  const bf16* p = baseT + n * stride;
  v16bf b;
#pragma unroll
  for (int v = 0; v < 8; ++v) {
    int k = kb + (v < 4 ? half * 8 + 2 * v : 16 + half * 8 + 2 * (v - 4));
    b[2 * v]     = p[k];
    b[2 * v + 1] = p[k + 1];
  }
  return b;
}

// ---------------- Tensor Data Mover (gfx1250) ----------------
#if defined(__gfx1250__) && __has_builtin(__builtin_amdgcn_tensor_load_to_lds)
#define USE_TDM 1
#else
#define USE_TDM 0
#endif

#if USE_TDM
typedef unsigned int tdm_v4u __attribute__((ext_vector_type(4)));
typedef int          tdm_v8i __attribute__((ext_vector_type(8)));
typedef int          tdm_v4i __attribute__((ext_vector_type(4)));

__device__ inline void wait_tensorcnt0() {
#if __has_builtin(__builtin_amdgcn_s_wait_tensorcnt)
  __builtin_amdgcn_s_wait_tensorcnt(0);
#else
  asm volatile("s_wait_tensorcnt 0x0" ::: "memory");
#endif
}

// DMA a 128x32 bf16 tile (rows x contiguous-K) from global into LDS.
// LDS pad: pad_interval = 16 DWORDs (one 64B row), pad_amount = 4 DWORDs
// -> effective LDS row stride = 40 bf16 elements (matches fragment loaders).
__device__ inline void tdm_load_a_tile(const bf16* gtile, unsigned lds_byte_off,
                                       int K, int Mtot) {
  const unsigned long long ga = (unsigned long long)(size_t)gtile;
  tdm_v4u g0;
  g0[0] = 1u;                                       // count=1, no gather
  g0[1] = lds_byte_off;                             // lds_addr (bytes)
  g0[2] = (unsigned)(ga & 0xffffffffull);           // global_addr[31:0]
  g0[3] = (unsigned)((ga >> 32) & 0x01ffffffull)    // global_addr[56:32]
        | 0x80000000u;                              // type=2 ("image")
  unsigned w[8];
  w[0] = (1u << 16)     // data_size = 2 bytes
       | (1u << 20)     // pad_enable
       | (3u << 22)     // pad_interval: 16 DWORDs
       | (3u << 25);    // pad_amount: 4 DWORDs
  w[1] = ((unsigned)K & 0xffffu) << 16;                               // tensor_dim0 lo16
  w[2] = (((unsigned)K >> 16) & 0xffffu) | (((unsigned)Mtot & 0xffffu) << 16);
  w[3] = (((unsigned)Mtot >> 16) & 0xffffu) | (32u << 16);            // tile_dim0 = 32
  w[4] = 128u;                                                        // tile_dim1 = 128
  w[5] = (unsigned)K;                                                 // dim0 stride lo32
  w[6] = 0u;
  w[7] = 0u;
  tdm_v8i g1;
#pragma unroll
  for (int i = 0; i < 8; ++i) g1[i] = (int)w[i];
  tdm_v4i z = {};
#if __clang_major__ >= 23
  tdm_v8i z8 = {};
  __builtin_amdgcn_tensor_load_to_lds(g0, g1, z, z, z8, 0);
#else
  __builtin_amdgcn_tensor_load_to_lds(g0, g1, z, z, 0);
#endif
}
#endif  // USE_TDM

// ---------------- weight / activation conversion ----------------
__global__ void f32_to_bf16_kernel(const float* __restrict__ in, bf16* __restrict__ out, size_t n) {
  size_t i = (size_t)blockIdx.x * blockDim.x + threadIdx.x;
  if (i < n) out[i] = (bf16)in[i];
}

// ---------------- RMSNorm: fp32 in -> bf16 out ----------------
__global__ __launch_bounds__(256)
void rmsnorm_kernel(const float* __restrict__ x, const float* __restrict__ w,
                    bf16* __restrict__ out, int ncols) {
  __shared__ float red[8];
  const size_t row = blockIdx.x;
  const float* xr = x + row * (size_t)ncols;
  float ss = 0.f;
  for (int i = threadIdx.x; i < ncols; i += 256) {
    float v = xr[i];
    ss += v * v;
  }
#pragma unroll
  for (int m = 16; m; m >>= 1) ss += __shfl_xor(ss, m, 32);
  if ((threadIdx.x & 31) == 0) red[threadIdx.x >> 5] = ss;
  __syncthreads();
  float tot = 0.f;
#pragma unroll
  for (int i = 0; i < 8; ++i) tot += red[i];
  const float inv = rsqrtf(tot / (float)ncols + kEPS);
  bf16* orow = out + row * (size_t)ncols;
  for (int i = threadIdx.x; i < ncols; i += 256)
    orow[i] = (bf16)(xr[i] * inv * w[i]);
}

// ---------------- GEMM: C[M,N] = A[M,K](bf16) * B[K,N](bf16) ----------------
// 128x128 block tile, 8 waves. Wave w: rows [(w&3)*32,+32), cols [(w>>2)*64,+64)
// -> 8 accumulators, 8 WMMAs per 32-wide k-step (B fragments reused 2x).
// Double-buffered LDS; A tile staged by TDM (wave 0) when available.
// OUT_BF16=1 : C bf16.  OUT_BF16=0 : C fp32 (+ optional fp32 residual).
constexpr int kLdsTile = 128 * 40;                      // elems per buffer
constexpr int kGemmLdsBytes = 4 * kLdsTile * 2;         // A[2] + Bt[2], bf16

template <int OUT_BF16>
__global__ __launch_bounds__(256)
void gemm_kernel(const bf16* __restrict__ A, const bf16* __restrict__ B,
                 void* __restrict__ C, const float* __restrict__ resid,
                 int M, int N, int K) {
  extern __shared__ __align__(16) bf16 smem[];
  bf16* Asm = smem;                  // A buffers at LDS offset 0 (TDM target)
  bf16* Btm = smem + 2 * kLdsTile;   // B^T buffers

  const int t = threadIdx.x;
  const int lane = t & 31, wave = t >> 5;
  const int m0 = blockIdx.y * 128, n0 = blockIdx.x * 128;
  const int mgrp = wave & 3;         // M rows [mgrp*32, +32)
  const int ngrp = wave >> 2;        // N cols [ngrp*64, +64)

  v8f acc0[4], acc1[4];
#pragma unroll
  for (int j = 0; j < 4; ++j) { acc0[j] = v8f{}; acc1[j] = v8f{}; }

  // B staging map: thread t -> k-row bk (0..31), 16 n-elems at bseg*16
  const int bk = t >> 3, bseg = t & 7;
#if !USE_TDM
  // A staging map: thread t -> row ar (0..127), 16 k-elems at aseg*16
  const int ar = t >> 1, aseg = t & 1;
#endif

  const int nk = K / 32;

  // ---- prologue: stage k-step 0 into buffer 0 ----
#if USE_TDM
  if (wave == 0) tdm_load_a_tile(A + (size_t)m0 * K, 0u, K, M);
#else
  {
    const bf16* asrc = A + (size_t)(m0 + ar) * K + aseg * 16;
    uint4 a0 = *(const uint4*)(asrc);
    uint4 a1 = *(const uint4*)(asrc + 8);
    *(uint4*)(Asm + ar * 40 + aseg * 16) = a0;
    *(uint4*)(Asm + ar * 40 + aseg * 16 + 8) = a1;
  }
#endif
  {
    const bf16* bsrc = B + (size_t)bk * N + n0 + bseg * 16;
    uint4 b0 = *(const uint4*)(bsrc);
    uint4 b1 = *(const uint4*)(bsrc + 8);
    const bf16* bp0 = (const bf16*)&b0;
    const bf16* bp1 = (const bf16*)&b1;
#pragma unroll
    for (int i = 0; i < 8; ++i) Btm[(bseg * 16 + i) * 40 + bk] = bp0[i];
#pragma unroll
    for (int i = 0; i < 8; ++i) Btm[(bseg * 16 + 8 + i) * 40 + bk] = bp1[i];
  }
#if USE_TDM
  if (wave == 0) wait_tensorcnt0();
#endif
  __syncthreads();

  // ---- main loop ----
  for (int i = 0; i < nk; ++i) {
    const int cur = i & 1, nxt = cur ^ 1;
    const bool have_next = (i + 1 < nk);
    const int k_next = (i + 1) * 32;

    uint4 br0 = {}, br1 = {};
#if !USE_TDM
    uint4 ar0 = {}, ar1 = {};
#endif
    if (have_next) {
#if USE_TDM
      if (wave == 0)
        tdm_load_a_tile(A + (size_t)m0 * K + k_next, (unsigned)(nxt * kLdsTile * 2), K, M);
#else
      const bf16* asrc = A + (size_t)(m0 + ar) * K + k_next + aseg * 16;
      ar0 = *(const uint4*)(asrc);
      ar1 = *(const uint4*)(asrc + 8);
#endif
      const bf16* bsrc = B + (size_t)(k_next + bk) * N + n0 + bseg * 16;
      br0 = *(const uint4*)(bsrc);
      br1 = *(const uint4*)(bsrc + 8);
      if (i + 4 < nk) {   // L2 prefetch a few k-steps ahead
        __builtin_prefetch(B + (size_t)((i + 4) * 32 + bk) * N + n0 + bseg * 16, 0, 0);
        __builtin_prefetch(A + (size_t)(m0 + (t >> 1)) * K + (i + 4) * 32, 0, 0);
      }
    }

    // compute from current buffers
    {
      const bf16* Ac = Asm + cur * kLdsTile;
      const bf16* Bc = Btm + cur * kLdsTile;
      v16bf a0 = load_a_frag(Ac + (mgrp * 32) * 40, 40, lane, 0);
      v16bf a1 = load_a_frag(Ac + (mgrp * 32 + 16) * 40, 40, lane, 0);
#pragma unroll
      for (int j = 0; j < 4; ++j) {
        v16bf b = load_b_frag(Bc + (ngrp * 64 + j * 16) * 40, 40, lane, 0);
        acc0[j] = wmma_bf16(a0, b, acc0[j]);
        acc1[j] = wmma_bf16(a1, b, acc1[j]);
      }
    }

    // store next-step staging
    if (have_next) {
      bf16* Bn = Btm + nxt * kLdsTile;
      const bf16* bp0 = (const bf16*)&br0;
      const bf16* bp1 = (const bf16*)&br1;
#pragma unroll
      for (int e = 0; e < 8; ++e) Bn[(bseg * 16 + e) * 40 + bk] = bp0[e];
#pragma unroll
      for (int e = 0; e < 8; ++e) Bn[(bseg * 16 + 8 + e) * 40 + bk] = bp1[e];
#if !USE_TDM
      bf16* An = Asm + nxt * kLdsTile;
      *(uint4*)(An + ar * 40 + aseg * 16) = ar0;
      *(uint4*)(An + ar * 40 + aseg * 16 + 8) = ar1;
#endif
    }
#if USE_TDM
    if (have_next && wave == 0) wait_tensorcnt0();
#endif
    __syncthreads();
  }

  // ---- epilogue: C layout -> M = r + 8*(lane>>4), N = lane&15 ----
  const int mrow0 = m0 + mgrp * 32 + (lane >> 4) * 8;
#pragma unroll
  for (int j = 0; j < 4; ++j) {
    const int nc = n0 + ngrp * 64 + j * 16 + (lane & 15);
#pragma unroll
    for (int r = 0; r < 8; ++r) {
      const size_t mA = (size_t)(mrow0 + r);
      const size_t mB = (size_t)(mrow0 + 16 + r);
      if (OUT_BF16) {
        ((bf16*)C)[mA * N + nc] = (bf16)acc0[j][r];
        ((bf16*)C)[mB * N + nc] = (bf16)acc1[j][r];
      } else {
        float v0 = acc0[j][r], v1 = acc1[j][r];
        if (resid) { v0 += resid[mA * N + nc]; v1 += resid[mB * N + nc]; }
        ((float*)C)[mA * N + nc] = v0;
        ((float*)C)[mB * N + nc] = v1;
      }
    }
  }
}

// ---------------- RoPE (in place on bf16 [rows][nheads*64]) ----------------
__global__ __launch_bounds__(256)
void rope_kernel(bf16* __restrict__ x, int nheads) {
  const size_t row = blockIdx.x;
  const int s = (int)(row % kS);
  const int npairs = nheads * 32;
  bf16* xr = x + row * (size_t)(nheads * kD);
  for (int idx = threadIdx.x; idx < npairs; idx += 256) {
    const int hh = idx >> 5, i = idx & 31;
    // inv_freq = 10000^(-2i/64)
    const float ang = (float)s * __expf(-((float)(2 * i) / 64.f) * 9.2103403719761836f);
    float c, sn;
    __sincosf(ang, &sn, &c);
    bf16* p = xr + hh * kD;
    const float x1 = (float)p[i], x2 = (float)p[i + 32];
    p[i]      = (bf16)(x1 * c - x2 * sn);
    p[i + 32] = (bf16)(x2 * c + x1 * sn);
  }
}

// ---------------- flash attention, causal, GQA ----------------
// grid: (S/64, H, B), block 128 (4 waves). wave w owns q-rows [w*16, w*16+16).
__global__ __launch_bounds__(128)
void attn_kernel(const bf16* __restrict__ Q, const bf16* __restrict__ K,
                 const bf16* __restrict__ V, bf16* __restrict__ O) {
  constexpr int LD = 72;  // even stride, 144B rows (16B aligned)
  __shared__ __align__(16) bf16 Qs[64 * LD];
  __shared__ __align__(16) bf16 Ks[64 * LD];
  __shared__ __align__(16) bf16 Vt[64 * LD];   // V transposed: [d][kv]
  __shared__ __align__(16) bf16 Ps[64 * LD];   // probs, 16 rows per wave

  const int t = threadIdx.x, lane = t & 31, wave = t >> 5;
  const int qtile = blockIdx.x, h = blockIdx.y, b = blockIdx.z;
  const int kvh = h >> 2;  // H/KVH = 4
  const size_t qrow0 = (size_t)b * kS + (size_t)qtile * 64;

  // load Q tile 64x64
  {
    const int r = t >> 1, seg = t & 1;
    const bf16* src = Q + (qrow0 + r) * (size_t)(kH * kD) + h * kD + seg * 32;
#pragma unroll
    for (int i = 0; i < 4; ++i)
      *(uint4*)(Qs + r * LD + seg * 32 + i * 8) = *(const uint4*)(src + i * 8);
  }

  float run_m[8], run_l[8];
  v8f o[4];
#pragma unroll
  for (int r = 0; r < 8; ++r) { run_m[r] = -1e30f; run_l[r] = 0.f; }
#pragma unroll
  for (int j = 0; j < 4; ++j) o[j] = v8f{};

  const float scale = 0.125f;  // 1/sqrt(64)
  const int half = lane >> 4;
  const int mloc = wave * 16 + half * 8;  // + r gives local q-row
  const int ntiles = qtile + 1;           // causal

  for (int tkv = 0; tkv < ntiles; ++tkv) {
    __syncthreads();
    {
      const int r = t >> 1, seg = t & 1;
      const size_t kvrow = (size_t)b * kS + (size_t)tkv * 64 + r;
      const bf16* ksrc = K + kvrow * (size_t)(kKVH * kD) + kvh * kD + seg * 32;
#pragma unroll
      for (int i = 0; i < 4; ++i)
        *(uint4*)(Ks + r * LD + seg * 32 + i * 8) = *(const uint4*)(ksrc + i * 8);
      const bf16* vsrc = V + kvrow * (size_t)(kKVH * kD) + kvh * kD + seg * 32;
#pragma unroll
      for (int i = 0; i < 4; ++i) {
        uint4 vv = *(const uint4*)(vsrc + i * 8);
        const bf16* vp = (const bf16*)&vv;
#pragma unroll
        for (int e = 0; e < 8; ++e) Vt[(seg * 32 + i * 8 + e) * LD + r] = vp[e];
      }
    }
    __syncthreads();

    // scores S = Q * K^T  (16 q-rows x 64 kv-cols per wave)
    v16bf qa0 = load_a_frag(Qs + wave * 16 * LD, LD, lane, 0);
    v16bf qa1 = load_a_frag(Qs + wave * 16 * LD, LD, lane, 32);
    v8f s[4];
#pragma unroll
    for (int j = 0; j < 4; ++j) {
      v16bf kb0 = load_b_frag(Ks + j * 16 * LD, LD, lane, 0);
      v16bf kb1 = load_b_frag(Ks + j * 16 * LD, LD, lane, 32);
      v8f z = {};
      z = wmma_bf16(qa0, kb0, z);
      z = wmma_bf16(qa1, kb1, z);
      s[j] = z;
    }

    // scale + causal mask
    const int qidx0 = qtile * 64 + mloc;
    const bool partial = (tkv == qtile);
#pragma unroll
    for (int j = 0; j < 4; ++j) {
      const int nidx = tkv * 64 + j * 16 + (lane & 15);
#pragma unroll
      for (int r = 0; r < 8; ++r) {
        float v = s[j][r] * scale;
        if (partial && nidx > qidx0 + r) v = -1e30f;
        s[j][r] = v;
      }
    }

    // per-row max over 64 kv columns (reduce across 16-lane half)
    float tmax[8];
#pragma unroll
    for (int r = 0; r < 8; ++r) {
      float m = s[0][r];
#pragma unroll
      for (int j = 1; j < 4; ++j) m = fmaxf(m, s[j][r]);
      m = fmaxf(m, __shfl_xor(m, 1, 32));
      m = fmaxf(m, __shfl_xor(m, 2, 32));
      m = fmaxf(m, __shfl_xor(m, 4, 32));
      m = fmaxf(m, __shfl_xor(m, 8, 32));
      tmax[r] = m;
    }

    float corr[8];
#pragma unroll
    for (int r = 0; r < 8; ++r) {
      const float nm = fmaxf(run_m[r], tmax[r]);
      corr[r] = __expf(run_m[r] - nm);
      run_m[r] = nm;
      run_l[r] *= corr[r];
    }
#pragma unroll
    for (int j = 0; j < 4; ++j)
#pragma unroll
      for (int r = 0; r < 8; ++r) o[j][r] *= corr[r];

    // exponentiate, accumulate row sums, spill P to LDS (C-layout -> memory)
    float rsum[8];
#pragma unroll
    for (int r = 0; r < 8; ++r) rsum[r] = 0.f;
#pragma unroll
    for (int j = 0; j < 4; ++j) {
      const int nc = j * 16 + (lane & 15);
#pragma unroll
      for (int r = 0; r < 8; ++r) {
        const float p = __expf(s[j][r] - run_m[r]);
        rsum[r] += p;
        Ps[(mloc + r) * LD + nc] = (bf16)p;
      }
    }
#pragma unroll
    for (int r = 0; r < 8; ++r) {
      float l = rsum[r];
      l += __shfl_xor(l, 1, 32);
      l += __shfl_xor(l, 2, 32);
      l += __shfl_xor(l, 4, 32);
      l += __shfl_xor(l, 8, 32);
      run_l[r] += l;
    }

    // O += P * V   (LDS ops are in-order within a wave)
    v16bf pa0 = load_a_frag(Ps + wave * 16 * LD, LD, lane, 0);
    v16bf pa1 = load_a_frag(Ps + wave * 16 * LD, LD, lane, 32);
#pragma unroll
    for (int j = 0; j < 4; ++j) {
      v16bf vb0 = load_b_frag(Vt + j * 16 * LD, LD, lane, 0);
      v16bf vb1 = load_b_frag(Vt + j * 16 * LD, LD, lane, 32);
      o[j] = wmma_bf16(pa0, vb0, o[j]);
      o[j] = wmma_bf16(pa1, vb1, o[j]);
    }
  }

  // normalize + write ctx (token-major [b*S+s][h*D+d] == reference reshape)
#pragma unroll
  for (int j = 0; j < 4; ++j) {
    const int nc = h * kD + j * 16 + (lane & 15);
#pragma unroll
    for (int r = 0; r < 8; ++r) {
      const size_t row = qrow0 + mloc + r;
      O[row * (size_t)(kH * kD) + nc] = (bf16)(o[j][r] / run_l[r]);
    }
  }
}

// ---------------- SwiGLU: act = silu(gate) * up ----------------
__global__ void swiglu_kernel(const bf16* __restrict__ g, const bf16* __restrict__ u,
                              bf16* __restrict__ out, size_t n) {
  size_t i = (size_t)blockIdx.x * blockDim.x + threadIdx.x;
  if (i < n) {
    const float gv = (float)g[i], uv = (float)u[i];
    const float sg = gv / (1.f + __expf(-gv));
    out[i] = (bf16)(sg * uv);
  }
}

// ---------------- host driver ----------------
extern "C" void kernel_launch(void* const* d_in, const int* in_sizes, int n_in,
                              void* d_out, int out_size, void* d_ws, size_t ws_size,
                              hipStream_t stream) {
  const float* hidden   = (const float*)d_in[0];
  // d_in[1] = attention_mask (causal mask applied analytically in-kernel)
  const float* norm1_w  = (const float*)d_in[2];
  const float* wq       = (const float*)d_in[3];
  const float* wk       = (const float*)d_in[4];
  const float* wv       = (const float*)d_in[5];
  const float* wo       = (const float*)d_in[6];
  const float* norm2_w  = (const float*)d_in[7];
  const float* w_gate   = (const float*)d_in[8];
  const float* w_up     = (const float*)d_in[9];
  const float* w_down   = (const float*)d_in[10];

  char* ws = (char*)d_ws;
  size_t off = 0;
  auto alloc = [&](size_t bytes) -> char* {
    char* p = ws + off;
    off += (bytes + 255) & ~(size_t)255;
    return p;
  };

  // bf16 weights
  bf16* wq_b = (bf16*)alloc((size_t)kHID * (kH * kD) * 2);
  bf16* wk_b = (bf16*)alloc((size_t)kHID * (kKVH * kD) * 2);
  bf16* wv_b = (bf16*)alloc((size_t)kHID * (kKVH * kD) * 2);
  bf16* wo_b = (bf16*)alloc((size_t)(kH * kD) * kHID * 2);
  bf16* wg_b = (bf16*)alloc((size_t)kHID * kFF * 2);
  bf16* wu_b = (bf16*)alloc((size_t)kHID * kFF * 2);
  bf16* wd_b = (bf16*)alloc((size_t)kFF * kHID * 2);
  // activations
  bf16*  hn   = (bf16*)alloc((size_t)kM * kHID * 2);
  bf16*  q_b  = (bf16*)alloc((size_t)kM * (kH * kD) * 2);
  bf16*  k_b  = (bf16*)alloc((size_t)kM * (kKVH * kD) * 2);
  bf16*  v_b  = (bf16*)alloc((size_t)kM * (kKVH * kD) * 2);
  bf16*  ctx  = (bf16*)alloc((size_t)kM * (kH * kD) * 2);
  float* h_f  = (float*)alloc((size_t)kM * kHID * 4);
  bf16*  h2   = (bf16*)alloc((size_t)kM * kHID * 2);
  bf16*  gate = (bf16*)alloc((size_t)kM * kFF * 2);
  bf16*  up   = (bf16*)alloc((size_t)kM * kFF * 2);
  bf16*  act  = (bf16*)alloc((size_t)kM * kFF * 2);

  auto cvt = [&](const float* src, bf16* dst, size_t n) {
    f32_to_bf16_kernel<<<dim3((unsigned)((n + 255) / 256)), 256, 0, stream>>>(src, dst, n);
  };
  cvt(wq,     wq_b, (size_t)kHID * (kH * kD));
  cvt(wk,     wk_b, (size_t)kHID * (kKVH * kD));
  cvt(wv,     wv_b, (size_t)kHID * (kKVH * kD));
  cvt(wo,     wo_b, (size_t)(kH * kD) * kHID);
  cvt(w_gate, wg_b, (size_t)kHID * kFF);
  cvt(w_up,   wu_b, (size_t)kHID * kFF);
  cvt(w_down, wd_b, (size_t)kFF * kHID);

  // 1) h = rmsnorm(hidden, norm1_w)  -> bf16
  rmsnorm_kernel<<<kM, 256, 0, stream>>>(hidden, norm1_w, hn, kHID);

  // 2) q/k/v projections
  gemm_kernel<1><<<dim3((kH * kD) / 128, kM / 128), 256, kGemmLdsBytes, stream>>>(
      hn, wq_b, q_b, nullptr, kM, kH * kD, kHID);
  gemm_kernel<1><<<dim3((kKVH * kD) / 128, kM / 128), 256, kGemmLdsBytes, stream>>>(
      hn, wk_b, k_b, nullptr, kM, kKVH * kD, kHID);
  gemm_kernel<1><<<dim3((kKVH * kD) / 128, kM / 128), 256, kGemmLdsBytes, stream>>>(
      hn, wv_b, v_b, nullptr, kM, kKVH * kD, kHID);

  // 3) RoPE on q and k (in place)
  rope_kernel<<<kM, 256, 0, stream>>>(q_b, kH);
  rope_kernel<<<kM, 256, 0, stream>>>(k_b, kKVH);

  // 4) causal GQA flash attention -> ctx (token-major)
  attn_kernel<<<dim3(kS / 64, kH, kB), 128, 0, stream>>>(q_b, k_b, v_b, ctx);

  // 5) attn_out = ctx @ wo ; h = residual + attn_out (fused)
  gemm_kernel<0><<<dim3(kHID / 128, kM / 128), 256, kGemmLdsBytes, stream>>>(
      ctx, wo_b, h_f, hidden, kM, kHID, kH * kD);

  // 6) h2 = rmsnorm(h, norm2_w)
  rmsnorm_kernel<<<kM, 256, 0, stream>>>(h_f, norm2_w, h2, kHID);

  // 7) gate/up projections
  gemm_kernel<1><<<dim3(kFF / 128, kM / 128), 256, kGemmLdsBytes, stream>>>(
      h2, wg_b, gate, nullptr, kM, kFF, kHID);
  gemm_kernel<1><<<dim3(kFF / 128, kM / 128), 256, kGemmLdsBytes, stream>>>(
      h2, wu_b, up, nullptr, kM, kFF, kHID);

  // 8) act = silu(gate) * up
  {
    size_t n = (size_t)kM * kFF;
    swiglu_kernel<<<dim3((unsigned)((n + 255) / 256)), 256, 0, stream>>>(gate, up, act, n);
  }

  // 9) out = h + act @ w_down (fused residual) -> d_out fp32
  gemm_kernel<0><<<dim3(kHID / 128, kM / 128), 256, kGemmLdsBytes, stream>>>(
      act, wd_b, (float*)d_out, h_f, kM, kHID, kFF);
}
